// CausalSelfAttention_59528246722829
// MI455X (gfx1250) — matrix operand order repaired and checked by reading, not compiled, not run on hardware
//
#include <hip/hip_runtime.h>

// ---------------------------------------------------------------------------
// CDNA5 (gfx1250) causal self-attention.
// bf16 WMMA for all matmuls; async Global->LDS staging (ASYNCcnt) for A/K
// tiles; Tensor Data Mover (TENSORcnt) for GEMM B tiles; double-buffered
// GEMM pipeline.  b=2, t=2048, e=1024, h=16, dh=64. wave32.
// ---------------------------------------------------------------------------

typedef __attribute__((ext_vector_type(16))) __bf16 v16bf;
typedef __attribute__((ext_vector_type(8)))  __bf16 v8bf;
typedef __attribute__((ext_vector_type(8)))  float  v8f;
typedef __attribute__((ext_vector_type(4)))  unsigned u32x4;
typedef __attribute__((ext_vector_type(8)))  unsigned u32x8;

union U16bf { v16bf v; v8bf h[2]; };

#define T_SEQ   2048
#define N_EMB   1024
#define N_HEADS 16
#define D_HEAD  64
#define M_ROWS  4096   // b * t

static __device__ __forceinline__ unsigned short f2bf(float f) {
    unsigned int u = __float_as_uint(f);
    unsigned int r = (u + 0x7FFFu + ((u >> 16) & 1u)) >> 16;  // RNE
    return (unsigned short)r;
}

// rope scale factor: cos(th)+sin(th), th = t * 10000^(-(d&31)/32)
static __device__ __forceinline__ float rope_scale(int t_idx, int d) {
    float inv = __powf(10000.0f, -((float)(d & 31)) * (1.0f / 32.0f));
    float th  = (float)t_idx * inv;
    return __cosf(th) + __sinf(th);
}

// --- CDNA5 async Global->LDS copy (16B per lane), tracked by ASYNCcnt ------
// Generic shared pointer low 32 bits == LDS byte offset (ISA aperture rule).
static __device__ __forceinline__ void async_copy_b128(const void* gsrc, void* ldst) {
    unsigned           lds = (unsigned)(unsigned long long)(uintptr_t)ldst;
    unsigned long long ga  = (unsigned long long)(uintptr_t)gsrc;
    asm volatile("global_load_async_to_lds_b128 %0, %1, off"
                 :: "v"(lds), "v"(ga) : "memory");
}
#define WAIT_ASYNC_0() asm volatile("s_wait_asynccnt 0x0" ::: "memory")
#define WAIT_ASYNC_2() asm volatile("s_wait_asynccnt 0x2" ::: "memory")

// --- CDNA5 Tensor Data Mover: 2D tile Global->LDS, tracked by TENSORcnt ----
// D# per ISA 8.3/8.4: group0 = {count, lds_addr, global_addr, type=2},
// group1 = {data_size, tensor dims, tile dims, dim0 stride}. 2-group form
// (VADDR2/3 = NULL) for tensors up to 2D. One issue moves the whole tile.
static __device__ __forceinline__ void tdm_load_tile_128x32_bf16(
    const unsigned short* gsrc, unsigned short* ldst, int K) {
    unsigned long long ga = (unsigned long long)(uintptr_t)gsrc;
    unsigned lds   = __builtin_amdgcn_readfirstlane((unsigned)(uintptr_t)ldst);
    unsigned ga_lo = __builtin_amdgcn_readfirstlane((unsigned)ga);
    unsigned ga_hi = __builtin_amdgcn_readfirstlane((unsigned)(ga >> 32));
    unsigned k     = __builtin_amdgcn_readfirstlane((unsigned)K);
    u32x4 g0;
    g0[0] = 1u;                                   // count=1, user descriptor
    g0[1] = lds;                                  // lds_addr (bytes)
    g0[2] = ga_lo;                                // global_addr[31:0]
    g0[3] = (ga_hi & 0x01FFFFFFu) | (2u << 30);   // global_addr[56:32] | type=2
    u32x8 g1;
    g1[0] = 1u << 16;                             // data_size = 2 bytes
    g1[1] = (k & 0xFFFFu) << 16;                  // tensor_dim0[15:0]
    g1[2] = (k >> 16) | (128u << 16);             // tensor_dim0[31:16] | tensor_dim1[15:0]
    g1[3] = 32u << 16;                            // tensor_dim1[31:16]=0 | tile_dim0=32
    g1[4] = 128u;                                 // tile_dim1=128 (tile_dim2=0)
    g1[5] = k;                                    // tensor_dim0_stride[31:0]
    g1[6] = 0u;                                   // stride[47:32] | dim1_stride lo
    g1[7] = 0u;
    asm volatile("tensor_load_to_lds %0, %1" :: "s"(g0), "s"(g1) : "memory");
}

// ------------------------------- f32 -> bf16 -------------------------------
__global__ __launch_bounds__(256) void cvt_bf16_kernel(const float* __restrict__ in,
                                                       unsigned short* __restrict__ out,
                                                       int n) {
    int i = blockIdx.x * 256 + threadIdx.x;
    if (i < n) out[i] = f2bf(in[i]);
}

// ------------------------------ WMMA GEMM ----------------------------------
// C[M,N] = A[M,K] * B[N,K]^T + bias, tiles 128x128x32, 8 waves per WG.
// A tile: per-thread async-to-LDS.  B tile: one TDM issue by wave 0.
// mode 0: Q (bias + rope, bf16)  1: K (bias + rope)  2: V (bias)
// mode 3: out-proj (bias, f32 -> d_out [M,N])
__global__ __launch_bounds__(256) void gemm_wmma_kernel(
    const unsigned short* __restrict__ A,   // [M,K] bf16
    const unsigned short* __restrict__ B,   // [N,K] bf16
    const float* __restrict__ bias,         // [N]
    unsigned short* __restrict__ outb,      // bf16 dest (modes 0..2)
    float* __restrict__ outf,               // f32 dest (mode 3)
    int M, int N, int K, int mode) {

    __shared__ unsigned short As[2][128 * 32];
    __shared__ unsigned short Bs[2][128 * 32];

    const int tid  = threadIdx.x;
    const int lane = tid & 31;
    const int wid  = tid >> 5;
    const int wm   = wid >> 2;      // 0..1  (64 rows each)
    const int wn   = wid & 3;       // 0..3  (32 cols each)
    const int bm   = blockIdx.y * 128;
    const int bn   = blockIdx.x * 128;
    const int lo16 = lane & 15;
    const bool hi  = lane >= 16;

    v8f acc[4][2];
    #pragma unroll
    for (int mt = 0; mt < 4; ++mt)
        #pragma unroll
        for (int nt = 0; nt < 2; ++nt) { v8f z = {}; acc[mt][nt] = z; }

    // per-thread A staging geometry: 2 chunks of 8 bf16
    const int off0 = tid * 8;
    const int off1 = (tid + 256) * 8;
    const int rA0 = off0 >> 5, cA0 = off0 & 31;
    const int rA1 = off1 >> 5, cA1 = off1 & 31;

    // issue stage 0
    {
        async_copy_b128(&A[(size_t)(bm + rA0) * K + cA0], &As[0][off0]);
        async_copy_b128(&A[(size_t)(bm + rA1) * K + cA1], &As[0][off1]);
        if (wid == 0)
            tdm_load_tile_128x32_bf16(&B[(size_t)bn * K], &Bs[0][0], K);
    }

    const int nk = K >> 5;
    for (int kb = 0; kb < nk; ++kb) {
        const int cur = kb & 1;
        if (kb + 1 < nk) {
            const int k1 = (kb + 1) << 5;
            async_copy_b128(&A[(size_t)(bm + rA0) * K + k1 + cA0], &As[cur ^ 1][off0]);
            async_copy_b128(&A[(size_t)(bm + rA1) * K + k1 + cA1], &As[cur ^ 1][off1]);
            if (wid == 0)
                tdm_load_tile_128x32_bf16(&B[(size_t)bn * K + k1], &Bs[cur ^ 1][0], K);
            WAIT_ASYNC_2();                          // in-order: stage kb's A done
            if (wid == 0) __builtin_amdgcn_s_wait_tensorcnt(1);  // stage kb's B done
        } else {
            WAIT_ASYNC_0();
            if (wid == 0) __builtin_amdgcn_s_wait_tensorcnt(0);
        }
        if (kb + 2 < nk) {      // far prefetch (global_prefetch_b8)
            __builtin_prefetch(&A[(size_t)(bm + (tid >> 1)) * K + ((kb + 2) << 5)], 0, 1);
            __builtin_prefetch(&B[(size_t)(bn + (tid >> 1)) * K + ((kb + 2) << 5)], 0, 1);
        }
        __syncthreads();

        // A fragments: lane<16 -> M=lane, K chunks {0..7},{16..23}; hi {8..15},{24..31}
        U16bf af[4];
        const int koA = hi ? 8 : 0;
        #pragma unroll
        for (int mt = 0; mt < 4; ++mt) {
            int row = wm * 64 + mt * 16 + lo16;
            af[mt].h[0] = *reinterpret_cast<const v8bf*>(&As[cur][row * 32 + koA]);
            af[mt].h[1] = *reinterpret_cast<const v8bf*>(&As[cur][row * 32 + 16 + koA]);
        }
        // B fragments: lane<16 -> N=lane, K chunks {0..7},{16..23}; hi {8..15},{24..31}
        U16bf bf_[2];
        const int koB = hi ? 8 : 0;
        #pragma unroll
        for (int nt = 0; nt < 2; ++nt) {
            int row = wn * 32 + nt * 16 + lo16;
            bf_[nt].h[0] = *reinterpret_cast<const v8bf*>(&Bs[cur][row * 32 + koB]);
            bf_[nt].h[1] = *reinterpret_cast<const v8bf*>(&Bs[cur][row * 32 + koB + 16]);
        }

        #pragma unroll
        for (int mt = 0; mt < 4; ++mt)
            #pragma unroll
            for (int nt = 0; nt < 2; ++nt)
                acc[mt][nt] = __builtin_amdgcn_wmma_f32_16x16x32_bf16(
                    false, af[mt].v, false, bf_[nt].v, (short)0, acc[mt][nt],
                    false, false);
        __syncthreads();   // buffer `cur` free before stage kb+2 overwrites it
    }

    // epilogue
    #pragma unroll
    for (int mt = 0; mt < 4; ++mt) {
        #pragma unroll
        for (int nt = 0; nt < 2; ++nt) {
            int col = bn + wn * 32 + nt * 16 + lo16;
            float bcol = bias[col];
            #pragma unroll
            for (int r = 0; r < 8; ++r) {
                int row = bm + wm * 64 + mt * 16 + r + (hi ? 8 : 0);
                float v = acc[mt][nt][r] + bcol;
                if (mode <= 2) {
                    int t_idx = row & (T_SEQ - 1);
                    int b_    = row >> 11;
                    int hh    = col >> 6;
                    int d     = col & (D_HEAD - 1);
                    if (mode <= 1) v *= rope_scale(t_idx, d);
                    outb[(((size_t)(b_ * N_HEADS + hh) * T_SEQ) + t_idx) * D_HEAD + d] = f2bf(v);
                } else {
                    outf[(size_t)row * N + col] = v;
                }
            }
        }
    }
}

// --------------------------- flash attention -------------------------------
// grid = (t/128, b*h); 8 waves, each owns 16 q rows. bf16 WMMA, f32 softmax.
__global__ __launch_bounds__(256) void attn_wmma_kernel(
    const unsigned short* __restrict__ Q,   // [b*h, t, 64] bf16 (rope'd)
    const unsigned short* __restrict__ Kt,  // [b*h, t, 64] bf16 (rope'd)
    const unsigned short* __restrict__ V,   // [b*h, t, 64] bf16
    unsigned short* __restrict__ Y) {       // [b, t, e] bf16

    __shared__ unsigned short Ks[128 * 64];       // [key][d]
    __shared__ unsigned short Vts[64 * 136];      // [d][key], padded stride 136
    __shared__ unsigned short Ps[8 * 16 * 32];    // per-wave P scratch [16][32]

    const int tid  = threadIdx.x;
    const int lane = tid & 31;
    const int w    = tid >> 5;
    const int lo16 = lane & 15;
    const bool hi  = lane >= 16;

    const int bh = blockIdx.y;          // b*16 + h
    const int b_ = bh >> 4;
    const int h_ = bh & 15;
    const int qb = blockIdx.x * 128;
    const int qw = qb + w * 16;         // this wave's first q row

    const size_t head_base = (size_t)bh * T_SEQ * D_HEAD;

    // Q A-fragments (persistent): d-base 0 and 32
    U16bf aq[2];
    {
        int qrow = qw + lo16;
        const unsigned short* qp = Q + head_base + (size_t)qrow * D_HEAD;
        #pragma unroll
        for (int f = 0; f < 2; ++f) {
            int d0 = f * 32 + (hi ? 8 : 0);
            aq[f].h[0] = *reinterpret_cast<const v8bf*>(qp + d0);
            aq[f].h[1] = *reinterpret_cast<const v8bf*>(qp + d0 + 16);
        }
    }

    v8f O[4];
    #pragma unroll
    for (int dt = 0; dt < 4; ++dt) { v8f z = {}; O[dt] = z; }
    float mrun[8], lsum[8];
    #pragma unroll
    for (int r = 0; r < 8; ++r) { mrun[r] = -1e30f; lsum[r] = 0.0f; }

    const int nkt = (qb >> 7) + 1;   // key tiles of 128, through the diagonal
    for (int kt = 0; kt < nkt; ++kt) {
        const int kbase = kt * 128;

        // ---- K tile [128][64]: async Global->LDS (4x b128 per thread) ----
        #pragma unroll
        for (int c = 0; c < 4; ++c) {
            int off = (tid + c * 256) * 8;        // 0..8184
            int kr  = off >> 6;                   // key row 0..127
            int dc  = off & 63;                   // d col
            async_copy_b128(&Kt[head_base + (size_t)(kbase + kr) * D_HEAD + dc], &Ks[off]);
        }
        // ---- V^T tile [64][128]: load + in-register transpose scatter ----
        #pragma unroll
        for (int c = 0; c < 4; ++c) {
            int off = (tid + c * 256) * 8;
            int kr  = off >> 6;
            int dc  = off & 63;
            const unsigned short* vp = &V[head_base + (size_t)(kbase + kr) * D_HEAD + dc];
            #pragma unroll
            for (int j = 0; j < 8; ++j)
                Vts[(dc + j) * 136 + kr] = vp[j];
        }
        WAIT_ASYNC_0();
        __syncthreads();

        if (kbase <= qw + 15) {
            for (int jj = 0; jj < 4; ++jj) {
                const int kstart = kbase + jj * 32;
                if (kstart > qw + 15) break;

                // ---- scores: S(16x32) = Q(16x64) . K^T, two 16-col tiles ----
                v8f S0, S1;
                {
                    const int koB = hi ? 8 : 0;
                    U16bf bk;
                    v8f z = {};
                    int krow = jj * 32 + lo16;             // keys kstart..+15
                    bk.h[0] = *reinterpret_cast<const v8bf*>(&Ks[krow * 64 + koB]);
                    bk.h[1] = *reinterpret_cast<const v8bf*>(&Ks[krow * 64 + koB + 16]);
                    S0 = __builtin_amdgcn_wmma_f32_16x16x32_bf16(false, aq[0].v, false, bk.v, (short)0, z, false, false);
                    bk.h[0] = *reinterpret_cast<const v8bf*>(&Ks[krow * 64 + 32 + koB]);
                    bk.h[1] = *reinterpret_cast<const v8bf*>(&Ks[krow * 64 + 32 + koB + 16]);
                    S0 = __builtin_amdgcn_wmma_f32_16x16x32_bf16(false, aq[1].v, false, bk.v, (short)0, S0, false, false);
                    krow += 16;                            // keys kstart+16..+31
                    bk.h[0] = *reinterpret_cast<const v8bf*>(&Ks[krow * 64 + koB]);
                    bk.h[1] = *reinterpret_cast<const v8bf*>(&Ks[krow * 64 + koB + 16]);
                    S1 = __builtin_amdgcn_wmma_f32_16x16x32_bf16(false, aq[0].v, false, bk.v, (short)0, z, false, false);
                    bk.h[0] = *reinterpret_cast<const v8bf*>(&Ks[krow * 64 + 32 + koB]);
                    bk.h[1] = *reinterpret_cast<const v8bf*>(&Ks[krow * 64 + 32 + koB + 16]);
                    S1 = __builtin_amdgcn_wmma_f32_16x16x32_bf16(false, aq[1].v, false, bk.v, (short)0, S1, false, false);
                }

                // ---- scale + causal mask + online softmax ----
                const int key0 = kstart + lo16;
                #pragma unroll
                for (int r = 0; r < 8; ++r) {
                    int qrow = qw + r + (hi ? 8 : 0);
                    float s0 = S0[r] * 0.125f;
                    float s1 = S1[r] * 0.125f;
                    if (key0 > qrow)      s0 = -1e30f;
                    if (key0 + 16 > qrow) s1 = -1e30f;

                    float mx = fmaxf(s0, s1);
                    #pragma unroll
                    for (int m = 1; m < 16; m <<= 1)
                        mx = fmaxf(mx, __shfl_xor(mx, m, 32));
                    float mn  = fmaxf(mrun[r], mx);
                    float cor = __expf(mrun[r] - mn);
                    float p0  = __expf(s0 - mn);
                    float p1  = __expf(s1 - mn);
                    float rs  = p0 + p1;
                    #pragma unroll
                    for (int m = 1; m < 16; m <<= 1)
                        rs += __shfl_xor(rs, m, 32);
                    lsum[r] = lsum[r] * cor + rs;
                    mrun[r] = mn;
                    #pragma unroll
                    for (int dt = 0; dt < 4; ++dt)
                        O[dt][r] *= cor;
                    // P into per-wave scratch (C layout -> [row][32])
                    int prow = r + (hi ? 8 : 0);
                    Ps[w * 512 + prow * 32 + lo16]      = f2bf(p0);
                    Ps[w * 512 + prow * 32 + 16 + lo16] = f2bf(p1);
                }

                // ---- O += P(16x32) . V(32x64) ----
                U16bf ap;
                {
                    int koA = hi ? 8 : 0;
                    ap.h[0] = *reinterpret_cast<const v8bf*>(&Ps[w * 512 + lo16 * 32 + koA]);
                    ap.h[1] = *reinterpret_cast<const v8bf*>(&Ps[w * 512 + lo16 * 32 + 16 + koA]);
                }
                const int koV = jj * 32 + (hi ? 8 : 0);
                #pragma unroll
                for (int dt = 0; dt < 4; ++dt) {
                    U16bf bv;
                    int dv = dt * 16 + lo16;
                    bv.h[0] = *reinterpret_cast<const v8bf*>(&Vts[dv * 136 + koV]);
                    bv.h[1] = *reinterpret_cast<const v8bf*>(&Vts[dv * 136 + koV + 16]);
                    O[dt] = __builtin_amdgcn_wmma_f32_16x16x32_bf16(
                        false, ap.v, false, bv.v, (short)0, O[dt], false, false);
                }
            }
        }
        __syncthreads();
    }

    // ---- finalize: Y[b, t, h*64+dv] = O / lsum ----
    #pragma unroll
    for (int dt = 0; dt < 4; ++dt) {
        int col = h_ * D_HEAD + dt * 16 + lo16;
        #pragma unroll
        for (int r = 0; r < 8; ++r) {
            int t_idx = qw + r + (hi ? 8 : 0);
            float v = O[dt][r] * (1.0f / lsum[r]);
            Y[((size_t)(b_ * T_SEQ + t_idx)) * N_EMB + col] = f2bf(v);
        }
    }
}

// ------------------------------- launcher ----------------------------------
extern "C" void kernel_launch(void* const* d_in, const int* in_sizes, int n_in,
                              void* d_out, int out_size, void* d_ws, size_t ws_size,
                              hipStream_t stream) {
    const float* x  = (const float*)d_in[0];
    const float* Wq = (const float*)d_in[1];
    const float* bq = (const float*)d_in[2];
    const float* Wk = (const float*)d_in[3];
    const float* bk = (const float*)d_in[4];
    const float* Wv = (const float*)d_in[5];
    const float* bv = (const float*)d_in[6];
    const float* Wo = (const float*)d_in[7];
    const float* bo = (const float*)d_in[8];
    float* out = (float*)d_out;

    char* ws = (char*)d_ws;
    const size_t SZ_X = (size_t)M_ROWS * N_EMB * 2;       // 8 MB
    const size_t SZ_W = (size_t)N_EMB * N_EMB * 2;        // 2 MB
    unsigned short* xb  = (unsigned short*)(ws);
    unsigned short* Wqb = (unsigned short*)(ws + SZ_X);
    unsigned short* Wkb = (unsigned short*)(ws + SZ_X + SZ_W);
    unsigned short* Wvb = (unsigned short*)(ws + SZ_X + 2 * SZ_W);
    unsigned short* Wob = (unsigned short*)(ws + SZ_X + 3 * SZ_W);
    unsigned short* Qb  = (unsigned short*)(ws + SZ_X + 4 * SZ_W);
    unsigned short* Kb  = (unsigned short*)(ws + 2 * SZ_X + 4 * SZ_W);
    unsigned short* Vb  = (unsigned short*)(ws + 3 * SZ_X + 4 * SZ_W);
    unsigned short* Yb  = (unsigned short*)(ws + 4 * SZ_X + 4 * SZ_W);

    const int nX = M_ROWS * N_EMB, nW = N_EMB * N_EMB;
    cvt_bf16_kernel<<<(nX + 255) / 256, 256, 0, stream>>>(x,  xb,  nX);
    cvt_bf16_kernel<<<(nW + 255) / 256, 256, 0, stream>>>(Wq, Wqb, nW);
    cvt_bf16_kernel<<<(nW + 255) / 256, 256, 0, stream>>>(Wk, Wkb, nW);
    cvt_bf16_kernel<<<(nW + 255) / 256, 256, 0, stream>>>(Wv, Wvb, nW);
    cvt_bf16_kernel<<<(nW + 255) / 256, 256, 0, stream>>>(Wo, Wob, nW);

    dim3 ggrid(N_EMB / 128, M_ROWS / 128);
    gemm_wmma_kernel<<<ggrid, 256, 0, stream>>>(xb, Wqb, bq, Qb, nullptr, M_ROWS, N_EMB, N_EMB, 0);
    gemm_wmma_kernel<<<ggrid, 256, 0, stream>>>(xb, Wkb, bk, Kb, nullptr, M_ROWS, N_EMB, N_EMB, 1);
    gemm_wmma_kernel<<<ggrid, 256, 0, stream>>>(xb, Wvb, bv, Vb, nullptr, M_ROWS, N_EMB, N_EMB, 2);

    dim3 agrid(T_SEQ / 128, 2 * N_HEADS);
    attn_wmma_kernel<<<agrid, 256, 0, stream>>>(Qb, Kb, Vb, Yb);

    gemm_wmma_kernel<<<ggrid, 256, 0, stream>>>(Yb, Wob, bo, nullptr, out, M_ROWS, N_EMB, N_EMB, 3);
}
